// Attention_53240414601843
// MI455X (gfx1250) — matrix operand order repaired and checked by reading, not compiled, not run on hardware
//
#include <hip/hip_runtime.h>
#include <hip/hip_bf16.h>

// ---------------------------------------------------------------------------
// MI455X (gfx1250) ViT attention block.
// ~77 GFLOP vs ~250MB traffic @ 23.3 TB/s -> matrix-core bound; every matmul
// stage runs on v_wmma_f32_16x16x32_bf16 (wave32). This revision:
//  - DPP16 row_xmask VALU reductions for the online softmax (replaces the
//    ds_bpermute + s_wait_dscnt round-trip chains seen in round-1 asm)
//  - global_load_async_to_lds_b128 (ASYNCcnt) for all LDS tile staging;
//    builtin signature on this toolchain: (v4i AS1*, v4i AS3*, imm, imm)
// ---------------------------------------------------------------------------

typedef __bf16 v16bf __attribute__((ext_vector_type(16)));
typedef float  v8f   __attribute__((ext_vector_type(8)));
typedef int    v4i_t __attribute__((ext_vector_type(4)));

union V16 {            // 32 bytes: one WMMA 16-bit A/B operand (16 bf16)
  uint4 u[2];
  v16bf v;
};

__device__ __forceinline__ unsigned short f2bf(float f) {
  unsigned int u = __float_as_uint(f);
  u += 0x7FFFu + ((u >> 16) & 1u);      // round-to-nearest-even
  return (unsigned short)(u >> 16);
}
__device__ __forceinline__ float bf2f(unsigned short h) {
  return __uint_as_float(((unsigned int)h) << 16);
}
__device__ __forceinline__ v8f v8f_zero() {
  v8f z;
#pragma unroll
  for (int i = 0; i < 8; ++i) z[i] = 0.0f;
  return z;
}

// ---- DPP16 row_xmask reduction helpers (VALU, no LDS round trips) ----------
#if __has_builtin(__builtin_amdgcn_update_dpp)
#define HAVE_DPP 1
template <int CTRL>
__device__ __forceinline__ float dppx(float v) {
  return __int_as_float(__builtin_amdgcn_update_dpp(
      0, __float_as_int(v), CTRL, 0xf, 0xf, true));
}
#endif

// ---- gfx1250 async global->LDS staging -------------------------------------
#if __has_builtin(__builtin_amdgcn_global_load_async_to_lds_b128)
#define HAVE_ASYNC 1
typedef __attribute__((address_space(1))) v4i_t* gp128_t;  // global v4i*
typedef __attribute__((address_space(3))) v4i_t* lp128_t;  // LDS v4i*
// INST_OFFSET applies to both global and LDS addresses (ISA 10.x), so a
// 32-byte chunk is two b128 ops with imm offsets 0 and 16.
template <int OFF>
__device__ __forceinline__ void async_ld128(const unsigned short* g,
                                            unsigned short* l) {
  __builtin_amdgcn_global_load_async_to_lds_b128(
      (gp128_t)const_cast<unsigned short*>(g), (lp128_t)l, OFF, 0);
}
__device__ __forceinline__ void wait_async0() {
#if __has_builtin(__builtin_amdgcn_s_wait_asynccnt)
  __builtin_amdgcn_s_wait_asynccnt(0);
#else
  asm volatile("s_wait_asynccnt 0x0" ::: "memory");
#endif
}
#endif

// ---------------------------------------------------------------------------
// f32 -> bf16 conversion, 4 elements/thread
// ---------------------------------------------------------------------------
__global__ void cvt_f32_bf16(const float* __restrict__ in,
                             unsigned short* __restrict__ out, int n4) {
  int i = blockIdx.x * blockDim.x + threadIdx.x;
  if (i >= n4) return;
  float4 f = ((const float4*)in)[i];
  ushort4 o;
  o.x = f2bf(f.x); o.y = f2bf(f.y); o.z = f2bf(f.z); o.w = f2bf(f.w);
  ((ushort4*)out)[i] = o;
}

// ---------------------------------------------------------------------------
// bf16 GEMM, K = 1024 fixed. C[M,N] = A[M,1024] * B[1024,N] + bias.
// 256 threads = 8 waves; block tile 128x128; wave tile 64x32 (4x2 WMMA tiles).
// ---------------------------------------------------------------------------
template <bool OUT_BF16>
__global__ void __launch_bounds__(256)
gemm_bf16_wmma(const unsigned short* __restrict__ A,
               const unsigned short* __restrict__ Bw,
               const float* __restrict__ bias,
               void* __restrict__ Cout, int M_valid, int N) {
  __shared__ unsigned short As[128 * 32];
  __shared__ unsigned short Bs[32 * 128];

  const int tid  = threadIdx.x;
  const int lane = tid & 31;
  const int wave = tid >> 5;
  const int m_blk = blockIdx.y * 128;
  const int n_blk = blockIdx.x * 128;
  const int wm = (wave >> 2) * 64;   // wave M offset: 0 / 64
  const int wn = (wave & 3) * 32;    // wave N offset: 0..96

  const int lrow = lane & 15;
  const int koff = (lane < 16) ? 0 : 8;   // ISA A-layout K chunk start
  const int hi8  = (lane < 16) ? 0 : 8;   // C/D layout row shift

  v8f acc[4][2];
#pragma unroll
  for (int i = 0; i < 4; ++i)
#pragma unroll
    for (int j = 0; j < 2; ++j) acc[i][j] = v8f_zero();

  // cooperative-load indexing
  const int ar = tid >> 1;           // A: 128 rows x 2 halves of 16 bf16
  const int ah = (tid & 1) * 16;
  const int br = tid >> 3;           // B: 32 rows x 8 chunks of 16 bf16
  const int bc = (tid & 7) * 16;

  for (int kk = 0; kk < 1024; kk += 32) {
    const int grow = m_blk + ar;
    const unsigned short* pA = A + (size_t)grow * 1024 + kk + ah;
    const unsigned short* pB = Bw + (size_t)(kk + br) * N + n_blk + bc;
    unsigned short* lA = &As[ar * 32 + ah];
    unsigned short* lB = &Bs[br * 128 + bc];
#if HAVE_ASYNC
    if (grow < M_valid) {
      async_ld128<0>(pA, lA);
      async_ld128<16>(pA, lA);
    } else {
      uint4 z = make_uint4(0, 0, 0, 0);
      *(uint4*)lA = z;
      *(uint4*)(lA + 8) = z;
    }
    async_ld128<0>(pB, lB);
    async_ld128<16>(pB, lB);
    wait_async0();
#else
    uint4 a0, a1;
    if (grow < M_valid) {
      const uint4* p = (const uint4*)pA;
      a0 = p[0]; a1 = p[1];
    } else {
      a0 = make_uint4(0, 0, 0, 0); a1 = a0;
    }
    *(uint4*)lA = a0;
    *(uint4*)(lA + 8) = a1;
    const uint4* q = (const uint4*)pB;
    *(uint4*)lB = q[0];
    *(uint4*)(lB + 8) = q[1];
#endif
    __syncthreads();

    V16 aop[4];
#pragma unroll
    for (int i = 0; i < 4; ++i) {
      int row = wm + i * 16 + lrow;
      aop[i].u[0] = *(const uint4*)&As[row * 32 + koff];
      aop[i].u[1] = *(const uint4*)&As[row * 32 + koff + 16];
    }
#pragma unroll
    for (int j = 0; j < 2; ++j) {
      V16 bop;   // lane = column n, elements over k in table order
#pragma unroll
      for (int e = 0; e < 16; ++e)
        ((unsigned short*)&bop)[e] = Bs[(koff + (e & 7) + ((e >> 3) << 4)) * 128 + wn + j * 16 + lrow];
#pragma unroll
      for (int i = 0; i < 4; ++i)
        acc[i][j] = __builtin_amdgcn_wmma_f32_16x16x32_bf16(
            false, aop[i].v, false, bop.v, (short)0, acc[i][j], false, false);
    }
    __syncthreads();
  }

  // epilogue: bias + store
#pragma unroll
  for (int j = 0; j < 2; ++j) {
    int col = n_blk + wn + j * 16 + lrow;
    float bv = bias[col];
#pragma unroll
    for (int i = 0; i < 4; ++i) {
#pragma unroll
      for (int r = 0; r < 8; ++r) {
        int row = m_blk + wm + i * 16 + r + hi8;
        if (row < M_valid) {
          float v = acc[i][j][r] + bv;
          if (OUT_BF16)
            ((unsigned short*)Cout)[(size_t)row * N + col] = f2bf(v);
          else
            ((float*)Cout)[(size_t)row * N + col] = v;
        }
      }
    }
  }
}

// ---------------------------------------------------------------------------
// RoPE + repack: qkv bf16 [5480,3072] -> Q [64,1376,64], K^T [64,64,1376],
// V [64,1376,64] (bf16, zero padded rows 1370..1375). prefix token n==0 skips
// RoPE; sin/cos indexed by (n-1, d).
// ---------------------------------------------------------------------------
__global__ void __launch_bounds__(256)
rope_pack(const unsigned short* __restrict__ qkv,
          const float* __restrict__ sinp, const float* __restrict__ cosp,
          unsigned short* __restrict__ Q, unsigned short* __restrict__ Kt,
          unsigned short* __restrict__ V) {
  int idx = blockIdx.x * blockDim.x + threadIdx.x;
  const int d = idx & 63;
  int t = idx >> 6;
  const int n = t % 1376;
  t /= 1376;
  const int h = t & 15;
  const int b = t >> 4;
  if (b >= 4) return;
  const int bh = b * 16 + h;

  unsigned short qo = 0, ko = 0, vo = 0;
  if (n < 1370) {
    const size_t rb = (size_t)(b * 1370 + n) * 3072 + h * 64;
    float qv = bf2f(qkv[rb + d]);
    float kv = bf2f(qkv[rb + 1024 + d]);
    float vv = bf2f(qkv[rb + 2048 + d]);
    if (n >= 1) {
      int d2 = (d < 32) ? d + 32 : d - 32;
      float sgn = (d < 32) ? -1.0f : 1.0f;
      float qp = bf2f(qkv[rb + d2]);
      float kp = bf2f(qkv[rb + 1024 + d2]);
      float sn = sinp[(size_t)(n - 1) * 64 + d];
      float cs = cosp[(size_t)(n - 1) * 64 + d];
      qv = qv * cs + sgn * qp * sn;
      kv = kv * cs + sgn * kp * sn;
    }
    qo = f2bf(qv); ko = f2bf(kv); vo = f2bf(vv);
  }
  Q[((size_t)bh * 1376 + n) * 64 + d]  = qo;
  Kt[((size_t)bh * 64 + d) * 1376 + n] = ko;
  V[((size_t)bh * 1376 + n) * 64 + d]  = vo;
}

// ---------------------------------------------------------------------------
// Flash attention. Block = 4 waves, each wave owns a 16-query tile; workgroup
// shares 32-key K^T / V tiles in LDS (async-staged). Key loop = 43 x 32 keys.
// Per step per wave: 4 WMMA for S = Q K^T, online softmax with DPP row_xmask
// reductions, P -> A-layout via per-wave LDS, 4 WMMA for O += P V.
// ---------------------------------------------------------------------------
__global__ void __launch_bounds__(128)
flash_attn(const unsigned short* __restrict__ Q,
           const unsigned short* __restrict__ Kt,
           const unsigned short* __restrict__ V,
           unsigned short* __restrict__ O) {
  __shared__ unsigned short Klds[64 * 32];      // [d][key]
  __shared__ unsigned short Vlds[32 * 64];      // [key][d]
  __shared__ unsigned short Plds[4 * 16 * 32];  // per-wave P transpose scratch

  const int tid  = threadIdx.x;
  const int lane = tid & 31;
  const int wave = tid >> 5;
  const int bh = blockIdx.x;
  const int b = bh >> 4, h = bh & 15;
  const int q0 = blockIdx.y * 64 + wave * 16;
  const int lrow = lane & 15;
  const int hi8  = (lane < 16) ? 0 : 8;
  const int koff = (lane < 16) ? 0 : 8;

  // Q A-operands (d 0..31, 32..63), loaded once
  int qrow = q0 + lrow;
  if (qrow > 1375) qrow = 1375;
  const unsigned short* qp = Q + ((size_t)bh * 1376 + qrow) * 64;
  V16 aq[2];
#pragma unroll
  for (int s = 0; s < 2; ++s) {
    aq[s].u[0] = *(const uint4*)(qp + s * 32 + koff);
    aq[s].u[1] = *(const uint4*)(qp + s * 32 + koff + 16);
  }

  float m[8], l[8];
  v8f acc[4];
#pragma unroll
  for (int r = 0; r < 8; ++r) { m[r] = -1e30f; l[r] = 0.0f; }
#pragma unroll
  for (int s = 0; s < 4; ++s) acc[s] = v8f_zero();

  const int kr = tid >> 1, kh = (tid & 1) * 16;  // K^T tile: 64 rows x 2 halves
  const int vr = tid >> 2, vq = (tid & 3) * 16;  // V tile: 32 rows x 4 quarters
  const float scale = 0.125f;                    // 1/sqrt(64)

  for (int kb = 0; kb < 1376; kb += 32) {
    // stage K^T / V tiles in LDS
    {
      const unsigned short* pk = Kt + ((size_t)bh * 64 + kr) * 1376 + kb + kh;
      const unsigned short* pv = V + ((size_t)bh * 1376 + kb + vr) * 64 + vq;
      unsigned short* lk = &Klds[kr * 32 + kh];
      unsigned short* lv = &Vlds[vr * 64 + vq];
#if HAVE_ASYNC
      async_ld128<0>(pk, lk);
      async_ld128<16>(pk, lk);
      async_ld128<0>(pv, lv);
      async_ld128<16>(pv, lv);
      wait_async0();
#else
      const uint4* gk = (const uint4*)pk;
      *(uint4*)lk = gk[0];
      *(uint4*)(lk + 8) = gk[1];
      const uint4* gv = (const uint4*)pv;
      *(uint4*)lv = gv[0];
      *(uint4*)(lv + 8) = gv[1];
#endif
    }
    __syncthreads();

    // S[16x32] = Q K^T  (two 16x16 subtiles, each two k=32 WMMA steps)
    v8f S[2];
#pragma unroll
    for (int sub = 0; sub < 2; ++sub) {
      v8f c = v8f_zero();
#pragma unroll
      for (int ds = 0; ds < 2; ++ds) {
        V16 bk;  // lane = key column n, elements over d in table order
#pragma unroll
        for (int e = 0; e < 16; ++e)
          ((unsigned short*)&bk)[e] = Klds[(ds * 32 + koff + (e & 7) + ((e >> 3) << 4)) * 32 + sub * 16 + lrow];
        c = __builtin_amdgcn_wmma_f32_16x16x32_bf16(
            false, aq[ds].v, false, bk.v, (short)0, c, false, false);
      }
      S[sub] = c;
    }

    // --- online softmax: rows r+hi8; 16 values/row across one lane half ---
    const bool v0ok = (kb + lrow) < 1370;
    const bool v1ok = (kb + 16 + lrow) < 1370;
    float s0[8], s1[8], mx[8];
#pragma unroll
    for (int r = 0; r < 8; ++r) {
      s0[r] = v0ok ? S[0][r] * scale : -1e30f;
      s1[r] = v1ok ? S[1][r] * scale : -1e30f;
      mx[r] = fmaxf(s0[r], s1[r]);
    }
    // row-of-16 max reduction
#if HAVE_DPP
#pragma unroll
    for (int r = 0; r < 8; ++r) {
      float v = mx[r];
      v = fmaxf(v, dppx<0x161>(v));   // row_xmask:1
      v = fmaxf(v, dppx<0x162>(v));   // row_xmask:2
      v = fmaxf(v, dppx<0x164>(v));   // row_xmask:4
      v = fmaxf(v, dppx<0x168>(v));   // row_xmask:8
      mx[r] = v;
    }
#else
#pragma unroll
    for (int o = 1; o < 16; o <<= 1)
#pragma unroll
      for (int r = 0; r < 8; ++r) mx[r] = fmaxf(mx[r], __shfl_xor(mx[r], o, 32));
#endif
    float P0[8], P1[8], ps[8];
#pragma unroll
    for (int r = 0; r < 8; ++r) {
      float mn = fmaxf(m[r], mx[r]);
      P0[r] = __expf(s0[r] - mn);
      P1[r] = __expf(s1[r] - mn);
      ps[r] = P0[r] + P1[r];
      float f = __expf(m[r] - mn);
      m[r] = mn;
      l[r] *= f;
#pragma unroll
      for (int s = 0; s < 4; ++s) acc[s][r] *= f;
    }
    // row-of-16 sum reduction
#if HAVE_DPP
#pragma unroll
    for (int r = 0; r < 8; ++r) {
      float v = ps[r];
      v += dppx<0x161>(v);
      v += dppx<0x162>(v);
      v += dppx<0x164>(v);
      v += dppx<0x168>(v);
      ps[r] = v;
    }
#else
#pragma unroll
    for (int o = 1; o < 16; o <<= 1)
#pragma unroll
      for (int r = 0; r < 8; ++r) ps[r] += __shfl_xor(ps[r], o, 32);
#endif
#pragma unroll
    for (int r = 0; r < 8; ++r) l[r] += ps[r];

    // P: D-layout -> A-layout via per-wave LDS (in-order per wave)
    unsigned short* pl = &Plds[wave * 16 * 32];
#pragma unroll
    for (int r = 0; r < 8; ++r) {
      pl[(r + hi8) * 32 + lrow]      = f2bf(P0[r]);
      pl[(r + hi8) * 32 + lrow + 16] = f2bf(P1[r]);
    }
    V16 pa;
    pa.u[0] = *(const uint4*)&pl[lrow * 32 + koff];
    pa.u[1] = *(const uint4*)&pl[lrow * 32 + koff + 16];

    // O += P V  (contraction over 32 keys, 4 d-segments of 16)
#pragma unroll
    for (int s = 0; s < 4; ++s) {
      V16 bv;  // lane = d column n, elements over keys in table order
#pragma unroll
      for (int e = 0; e < 16; ++e)
        ((unsigned short*)&bv)[e] = Vlds[(koff + (e & 7) + ((e >> 3) << 4)) * 64 + s * 16 + lrow];
      acc[s] = __builtin_amdgcn_wmma_f32_16x16x32_bf16(
          false, pa.v, false, bv.v, (short)0, acc[s], false, false);
    }
    __syncthreads();
  }

  // normalize + store bf16 to [b*1370+n, h*64+d]
#pragma unroll
  for (int s = 0; s < 4; ++s) {
#pragma unroll
    for (int r = 0; r < 8; ++r) {
      int qi = q0 + r + hi8;
      if (qi < 1370) {
        int col = h * 64 + s * 16 + lrow;
        O[(size_t)(b * 1370 + qi) * 1024 + col] = f2bf(acc[s][r] / l[r]);
      }
    }
  }
}

// ---------------------------------------------------------------------------
// launcher
// ---------------------------------------------------------------------------
extern "C" void kernel_launch(void* const* d_in, const int* in_sizes, int n_in,
                              void* d_out, int out_size, void* d_ws, size_t ws_size,
                              hipStream_t stream) {
  const float* x      = (const float*)d_in[0];
  const float* sinp   = (const float*)d_in[1];
  const float* cosp   = (const float*)d_in[2];
  const float* w_qkv  = (const float*)d_in[3];
  const float* b_qkv  = (const float*)d_in[4];
  const float* w_proj = (const float*)d_in[5];
  const float* b_proj = (const float*)d_in[6];
  float* out = (float*)d_out;

  char* ws = (char*)d_ws;
  size_t off = 0;
  auto alloc = [&](size_t bytes) -> void* {
    void* p = ws + off;
    off = (off + bytes + 255) & ~(size_t)255;
    return p;
  };
  unsigned short* Xbf     = (unsigned short*)alloc(5480ull * 1024 * 2);
  unsigned short* Wqkvbf  = (unsigned short*)alloc(1024ull * 3072 * 2);
  unsigned short* Wprojbf = (unsigned short*)alloc(1024ull * 1024 * 2);
  unsigned short* QKVbf   = (unsigned short*)alloc(5480ull * 3072 * 2);
  unsigned short* Qb      = (unsigned short*)alloc(64ull * 1376 * 64 * 2);
  unsigned short* Ktb     = (unsigned short*)alloc(64ull * 1376 * 64 * 2);
  unsigned short* Vb      = (unsigned short*)alloc(64ull * 1376 * 64 * 2);
  unsigned short* Ob      = (unsigned short*)alloc(5480ull * 1024 * 2);

  // f32 -> bf16 conversions
  cvt_f32_bf16<<<5480, 256, 0, stream>>>(x, Xbf, 5611520 / 4);
  cvt_f32_bf16<<<3072, 256, 0, stream>>>(w_qkv, Wqkvbf, 3145728 / 4);
  cvt_f32_bf16<<<1024, 256, 0, stream>>>(w_proj, Wprojbf, 1048576 / 4);

  // QKV projection: [5480,1024] x [1024,3072] + b_qkv -> bf16
  gemm_bf16_wmma<true><<<dim3(24, 43), 256, 0, stream>>>(
      Xbf, Wqkvbf, b_qkv, QKVbf, 5480, 3072);

  // RoPE + head-major repack (Q, K^T, V), zero-padded to N=1376
  rope_pack<<<22016, 256, 0, stream>>>(QKVbf, sinp, cosp, Qb, Ktb, Vb);

  // attention: grid (B*H = 64, query tiles of 64 -> 22)
  flash_attn<<<dim3(64, 22), 128, 0, stream>>>(Qb, Ktb, Vb, Ob);

  // output projection: [5480,1024] x [1024,1024] + b_proj -> f32 d_out
  gemm_bf16_wmma<false><<<dim3(8, 43), 256, 0, stream>>>(
      Ob, Wprojbf, b_proj, out, 5480, 1024);
}
